// TPRUEncoder_13030930776430
// MI455X (gfx1250) — matrix operand order repaired and checked by reading, not compiled, not run on hardware
//
#include <hip/hip_runtime.h>
#include <hip/hip_bf16.h>
#include <math.h>

typedef unsigned short u16t;
typedef unsigned int   u32t;
typedef __attribute__((ext_vector_type(16))) __bf16 v16bf;
typedef __attribute__((ext_vector_type(8)))  float  v8f;

#define BB 256   // batch
#define SS 512   // sequence length
#define DD 1024  // D
#define NN 2048  // N
#define LPAD 40  // LDS row stride in bf16 (32 data + 8 pad) -> 80B rows, 16B aligned

__device__ __forceinline__ u16t f2bf(float f) {
  u32t u = __float_as_uint(f);
  u32t r = (u + 0x7FFFu + ((u >> 16) & 1u)) >> 16;   // round-to-nearest-even
  return (u16t)r;
}
__device__ __forceinline__ float bf2f(u16t h) {
  return __uint_as_float(((u32t)h) << 16);
}

// ---- WMMA fragment gathers from LDS (16x16x32 bf16 layouts, wave32) ----
// A (16x32): lanes 0-15 rows M=0..15 with K in {0..7,16..23}; lanes 16-31 K+8.
__device__ __forceinline__ v16bf frag_a(const u16t* ls, int mrow) {
  int lane = threadIdx.x & 31;
  int row  = mrow + (lane & 15);
  int hi   = (lane >> 4) & 1;
  union { uint4 u[2]; v16bf v; } f;
  f.u[0] = *(const uint4*)(ls + row * LPAD + hi * 8);
  f.u[1] = *(const uint4*)(ls + row * LPAD + 16 + hi * 8);
  return f.v;
}
// B (32x16): lane = N (lo/hi half), K in {0..7,16..23} (lanes<16) / K+8 (lanes>=16).
// LDS holds Bt row-major (N x K) so this is 2 contiguous b128 reads.
__device__ __forceinline__ v16bf frag_b(const u16t* ls, int nrow) {
  int lane = threadIdx.x & 31;
  int row  = nrow + (lane & 15);
  int hi   = (lane >> 4) & 1;
  union { uint4 u[2]; v16bf v; } f;
  f.u[0] = *(const uint4*)(ls + row * LPAD + hi * 8);
  f.u[1] = *(const uint4*)(ls + row * LPAD + 16 + hi * 8);
  return f.v;
}

#define WMMA_BF16(Af, Bf, Cf) \
  __builtin_amdgcn_wmma_f32_16x16x32_bf16(false, (Af), false, (Bf), (short)0, (Cf), false, false)

// ---- CDNA5 async global -> LDS copy (no VGPR round trip, ASYNCcnt) ----
// Per-lane: 16B from global addr to LDS byte offset (low 32 bits of generic ptr).
__device__ __forceinline__ void async_ld16(u32t lds_off, const u16t* g) {
  asm volatile("global_load_async_to_lds_b128 %0, %1, off"
               : : "v"(lds_off), "v"(g) : "memory");
}

// Double-buffered, async-pipelined K-loop: C(64x64) += A(m0.., :) * Bt(n0.., :)^T
// A and Bt are row-major bf16 with leading dimension == K. 128 threads / 4 waves,
// wave (wm,wn) owns a 32x32 sub-tile = 2x2 WMMA accumulators.
__device__ __forceinline__ void mm_tiles(const u16t* __restrict__ A, long m0,
                                         const u16t* __restrict__ Bt, long n0,
                                         int K,
                                         u16t* lsA0, u16t* lsA1,
                                         u16t* lsB0, u16t* lsB1,
                                         int wm, int wn, v8f (&acc)[2][2]) {
  int tid = (int)threadIdx.x;
  // two 16B chunks per thread per 64x32 tile
  int c0 = tid, c1 = tid + 128;
  int r0 = c0 >> 2, col0 = (c0 & 3) * 8;
  int r1 = c1 >> 2, col1 = (c1 & 3) * 8;
  const u16t* gA0 = A + (m0 + r0) * (long)K + col0;
  const u16t* gA1 = A + (m0 + r1) * (long)K + col1;
  const u16t* gB0 = Bt + (n0 + r0) * (long)K + col0;
  const u16t* gB1 = Bt + (n0 + r1) * (long)K + col1;
  u32t o0 = (u32t)(r0 * LPAD + col0) * 2u;
  u32t o1 = (u32t)(r1 * LPAD + col1) * 2u;
  u32t bA0 = (u32t)(uintptr_t)lsA0, bA1 = (u32t)(uintptr_t)lsA1;
  u32t bB0 = (u32t)(uintptr_t)lsB0, bB1 = (u32t)(uintptr_t)lsB1;

  // prologue: fetch first K-slice into buffer 0
  async_ld16(bA0 + o0, gA0); async_ld16(bA0 + o1, gA1);
  async_ld16(bB0 + o0, gB0); async_ld16(bB0 + o1, gB1);

  int nk = K >> 5;
  for (int kk = 0; kk < nk; ++kk) {
    const u16t* la = (kk & 1) ? lsA1 : lsA0;
    const u16t* lb = (kk & 1) ? lsB1 : lsB0;
    if (kk + 1 < nk) {
      long ko = (long)(kk + 1) * 32;
      u32t dA = (kk & 1) ? bA0 : bA1;
      u32t dB = (kk & 1) ? bB0 : bB1;
      async_ld16(dA + o0, gA0 + ko); async_ld16(dA + o1, gA1 + ko);
      async_ld16(dB + o0, gB0 + ko); async_ld16(dB + o1, gB1 + ko);
      // 8 outstanding; <=4 left means current buffer's group landed (in-order)
      asm volatile("s_wait_asynccnt 0x4" ::: "memory");
    } else {
      asm volatile("s_wait_asynccnt 0x0" ::: "memory");
    }
    __syncthreads();
    v16bf a0 = frag_a(la, wm), a1 = frag_a(la, wm + 16);
    v16bf b0 = frag_b(lb, wn), b1 = frag_b(lb, wn + 16);
    acc[0][0] = WMMA_BF16(a0, b0, acc[0][0]);
    acc[0][1] = WMMA_BF16(a0, b1, acc[0][1]);
    acc[1][0] = WMMA_BF16(a1, b0, acc[1][0]);
    acc[1][1] = WMMA_BF16(a1, b1, acc[1][1]);
    __syncthreads();  // all reads done before this buffer is refilled
  }
}

#define DECL_LDS()                 \
  __shared__ u16t lsA0[64 * LPAD]; \
  __shared__ u16t lsA1[64 * LPAD]; \
  __shared__ u16t lsB0[64 * LPAD]; \
  __shared__ u16t lsB1[64 * LPAD]

// ---------------- generic NT GEMM (bf16 in, bf16 out) for precompute --------
__global__ void __launch_bounds__(128)
gemm_nt_bf16(const u16t* __restrict__ A, const u16t* __restrict__ Bt,
             u16t* __restrict__ C, int K, int ldc) {
  DECL_LDS();
  long m0 = (long)blockIdx.x * 64;
  long n0 = (long)blockIdx.y * 64;
  int wave = threadIdx.x >> 5;
  int wm = (wave & 1) * 32, wn = (wave >> 1) * 32;
  v8f acc[2][2] = {};
  mm_tiles(A, m0, Bt, n0, K, lsA0, lsA1, lsB0, lsB1, wm, wn, acc);
  int lane = threadIdx.x & 31;
  int hi   = (lane >> 4) & 1;
#pragma unroll
  for (int i = 0; i < 2; ++i)
#pragma unroll
    for (int j = 0; j < 2; ++j) {
      long nc = n0 + wn + j * 16 + (lane & 15);
      long mb = m0 + wm + i * 16 + hi * 8;
#pragma unroll
      for (int r = 0; r < 8; ++r)
        C[(mb + r) * (long)ldc + nc] = f2bf(acc[i][j][r]);
    }
}

// --------- step kernel F: num = (relu(UT@btT) + relu(uw*xt))^2 ------------
// stores numT (B x N, bf16) and per-batch column sums via atomics.
__global__ void __launch_bounds__(128)
step_f(const u16t* __restrict__ UT, const u16t* __restrict__ btbf,
       const float* __restrict__ uw, const float* __restrict__ seq, int t,
       u16t* __restrict__ numT, float* __restrict__ colsum) {
  DECL_LDS();
  long n0 = (long)blockIdx.x * 64;  // over N
  long b0 = (long)blockIdx.y * 64;  // over B
  int wave = threadIdx.x >> 5;
  int wm = (wave & 1) * 32, wn = (wave >> 1) * 32;
  v8f acc[2][2] = {};
  mm_tiles(UT, n0, btbf, b0, DD, lsA0, lsA1, lsB0, lsB1, wm, wn, acc);
  int lane = threadIdx.x & 31;
  int hi   = (lane >> 4) & 1;
#pragma unroll
  for (int j = 0; j < 2; ++j) {
    long bcol = b0 + wn + j * 16 + (lane & 15);
    float x = seq[bcol * SS + t];
#pragma unroll
    for (int i = 0; i < 2; ++i) {
      long nb = n0 + wm + i * 16 + hi * 8;
      float s = 0.f;
      union { u16t h[8]; uint4 u; } pk;
#pragma unroll
      for (int r = 0; r < 8; ++r) {
        float fbt = fmaxf(acc[i][j][r], 0.f);
        float fxt = fmaxf(uw[nb + r] * x, 0.f);
        float v = fbt + fxt;
        v *= v;
        s += v;
        pk.h[r] = f2bf(v);
      }
      *(uint4*)(numT + bcol * NN + nb) = pk.u;  // 8 contiguous bf16
      atomicAdd(colsum + bcol, s);
    }
  }
}

// --------- step kernel G: b = (R@num)/colsum ; g = sigmoid(Wb@btT + wx*xt);
//           bt_new = g*tanh(b) + (1-g)*bt_old -----------------------------
__global__ void __launch_bounds__(128)
step_g(const u16t* __restrict__ Rb, const u16t* __restrict__ numT,
       const u16t* __restrict__ Wbb, const u16t* __restrict__ btbf_old,
       const float* __restrict__ bt_old, const float* __restrict__ colsum,
       const float* __restrict__ Wx, const float* __restrict__ seq, int t,
       float* __restrict__ bt_new, u16t* __restrict__ btbf_new) {
  DECL_LDS();
  long d0 = (long)blockIdx.x * 64;  // over D
  long b0 = (long)blockIdx.y * 64;  // over B
  int wave = threadIdx.x >> 5;
  int wm = (wave & 1) * 32, wn = (wave >> 1) * 32;
  v8f acc1[2][2] = {}, acc2[2][2] = {};
  // phase 1: R @ num   (K = NN)
  mm_tiles(Rb, d0, numT, b0, NN, lsA0, lsA1, lsB0, lsB1, wm, wn, acc1);
  // phase 2: Wb @ btT  (K = DD)
  mm_tiles(Wbb, d0, btbf_old, b0, DD, lsA0, lsA1, lsB0, lsB1, wm, wn, acc2);
  int lane = threadIdx.x & 31;
  int hi   = (lane >> 4) & 1;
#pragma unroll
  for (int j = 0; j < 2; ++j) {
    long bcol = b0 + wn + j * 16 + (lane & 15);
    float x  = seq[bcol * SS + t];
    float rs = 1.f / (colsum[bcol] + 1e-6f);
#pragma unroll
    for (int i = 0; i < 2; ++i) {
      long db = d0 + wm + i * 16 + hi * 8;
      union { u16t h[8]; uint4 u; } pk;
#pragma unroll
      for (int r = 0; r < 8; ++r) {
        long d = db + r;
        float bm = acc1[i][j][r] * rs;
        float hh = acc2[i][j][r] + Wx[d] * x;
        float g  = 1.f / (1.f + expf(-hh));
        float bo = bt_old[bcol * DD + d];
        float bn = g * tanhf(bm) + (1.f - g) * bo;
        bt_new[bcol * DD + d] = bn;
        pk.h[r] = f2bf(bn);
      }
      *(uint4*)(btbf_new + bcol * DD + db) = pk.u;
    }
  }
}

// ---------------- small helper kernels ----------------
__global__ void cvt_f32_bf16(const float* __restrict__ s, u16t* __restrict__ d, int n) {
  int i = blockIdx.x * blockDim.x + threadIdx.x;
  if (i < n) d[i] = f2bf(s[i]);
}
__global__ void transpose_cvt(const float* __restrict__ V, u16t* __restrict__ Vt) {
  long i = (long)blockIdx.x * blockDim.x + threadIdx.x;  // over NN*DD
  if (i < (long)NN * DD) {
    long n = i / DD, d = i - n * DD;
    Vt[i] = f2bf(V[d * NN + n]);
  }
}
__global__ void uw_kernel(const u16t* __restrict__ UT, const float* __restrict__ W,
                          float* __restrict__ uw) {
  int n = blockIdx.x * blockDim.x + threadIdx.x;
  if (n < NN) {
    float s = 0.f;
    const u16t* row = UT + (long)n * DD;
    for (int d = 0; d < DD; ++d) s += bf2f(row[d]) * W[d];
    uw[n] = s;
  }
}
__global__ void zero_f32(float* __restrict__ p, int n) {
  int i = blockIdx.x * blockDim.x + threadIdx.x;
  if (i < n) p[i] = 0.f;
}
__global__ void zero_u16(u16t* __restrict__ p, int n) {
  int i = blockIdx.x * blockDim.x + threadIdx.x;
  if (i < n) p[i] = 0;
}
__global__ void copy_out(const float* __restrict__ s, float* __restrict__ d, int n) {
  int i = blockIdx.x * blockDim.x + threadIdx.x;
  if (i < n) d[i] = s[i];
}

extern "C" void kernel_launch(void* const* d_in, const int* in_sizes, int n_in,
                              void* d_out, int out_size, void* d_ws, size_t ws_size,
                              hipStream_t stream) {
  const float* seq = (const float*)d_in[0];  // (B,S)
  const float* Wu  = (const float*)d_in[1];  // (D,D)
  const float* Wr  = (const float*)d_in[2];  // (D,D)
  const float* Wb  = (const float*)d_in[3];  // (D,D)
  const float* W   = (const float*)d_in[4];  // (D,1)
  const float* Wx  = (const float*)d_in[5];  // (D,1)
  const float* V   = (const float*)d_in[6];  // (D,N)

  char* ws = (char*)d_ws;
  size_t off = 0;
  auto alloc = [&](size_t bytes) -> void* {
    void* p = ws + off;
    off += (bytes + 255) & ~(size_t)255;
    return p;
  };
  u16t* Wubf  = (u16t*)alloc((size_t)DD * DD * 2);
  u16t* Wrbf  = (u16t*)alloc((size_t)DD * DD * 2);
  u16t* Wbbf  = (u16t*)alloc((size_t)DD * DD * 2);
  u16t* Vtbf  = (u16t*)alloc((size_t)NN * DD * 2);  // V transposed (N x D)
  u16t* UTbf  = (u16t*)alloc((size_t)NN * DD * 2);  // (Wu@V)^T  (N x D)
  u16t* Rbf   = (u16t*)alloc((size_t)DD * NN * 2);  // Wr@V      (D x N)
  float* uw   = (float*)alloc((size_t)NN * 4);      // UT @ W    (N)
  u16t* numT  = (u16t*)alloc((size_t)BB * NN * 2);  // num^T     (B x N)
  float* cs   = (float*)alloc((size_t)BB * 4);      // column sums
  float* btA  = (float*)alloc((size_t)BB * DD * 4);
  float* btB  = (float*)alloc((size_t)BB * DD * 4);
  u16t* bfA   = (u16t*)alloc((size_t)BB * DD * 2);
  u16t* bfB   = (u16t*)alloc((size_t)BB * DD * 2);
  (void)ws_size; (void)in_sizes; (void)n_in;

  // --- one-time precompute ---
  cvt_f32_bf16<<<(DD * DD + 255) / 256, 256, 0, stream>>>(Wu, Wubf, DD * DD);
  cvt_f32_bf16<<<(DD * DD + 255) / 256, 256, 0, stream>>>(Wr, Wrbf, DD * DD);
  cvt_f32_bf16<<<(DD * DD + 255) / 256, 256, 0, stream>>>(Wb, Wbbf, DD * DD);
  transpose_cvt<<<(int)(((long)NN * DD + 255) / 256), 256, 0, stream>>>(V, Vtbf);
  // UT(n,d) = sum_k V(k,n)*Wu(d,k) : A=Vt (NxD), Bt=Wu (DxD)
  gemm_nt_bf16<<<dim3(NN / 64, DD / 64), 128, 0, stream>>>(Vtbf, Wubf, UTbf, DD, DD);
  // R(d,n) = sum_k Wr(d,k)*V(k,n) : A=Wr (DxD), Bt=Vt (NxD)
  gemm_nt_bf16<<<dim3(DD / 64, NN / 64), 128, 0, stream>>>(Wrbf, Vtbf, Rbf, DD, NN);
  uw_kernel<<<NN / 256, 256, 0, stream>>>(UTbf, W, uw);
  zero_f32<<<(BB * DD + 255) / 256, 256, 0, stream>>>(btA, BB * DD);
  zero_u16<<<(BB * DD + 255) / 256, 256, 0, stream>>>(bfA, BB * DD);

  // --- recurrent scan ---
  for (int t = 0; t < SS; ++t) {
    float* bt_cur = (t & 1) ? btB : btA;
    float* bt_nxt = (t & 1) ? btA : btB;
    u16t* bf_cur  = (t & 1) ? bfB : bfA;
    u16t* bf_nxt  = (t & 1) ? bfA : bfB;
    zero_f32<<<1, 256, 0, stream>>>(cs, BB);
    step_f<<<dim3(NN / 64, BB / 64), 128, 0, stream>>>(UTbf, bf_cur, uw, seq, t, numT, cs);
    step_g<<<dim3(DD / 64, BB / 64), 128, 0, stream>>>(Rbf, numT, Wbbf, bf_cur, bt_cur,
                                                       cs, Wx, seq, t, bt_nxt, bf_nxt);
  }
  // after 512 steps (even count) the final state is in btA
  copy_out<<<(BB * DD + 255) / 256, 256, 0, stream>>>(btA, (float*)d_out, BB * DD);
}
